// MultiHeadSelfattention_1468878815456
// MI455X (gfx1250) — matrix operand order repaired and checked by reading, not compiled, not run on hardware
//
#include <hip/hip_runtime.h>
#include <hip/hip_bf16.h>

// ---------------------------------------------------------------------------
// MI455X (gfx1250) multi-head self-attention forward.
// - v_wmma_f32_16x16x32_bf16 (wave32 WMMA), bf16 operands, fp32 accumulation
// - flash-style online softmax
// - GLOBAL_LOAD_ASYNC_TO_LDS_B128 staging (ASYNCcnt) via gfx1250 builtins
// - global_prefetch for next-K-step tiles in the fp32 GEMM loaders
// ---------------------------------------------------------------------------

typedef __attribute__((ext_vector_type(16))) __bf16 v16bf;
typedef __attribute__((ext_vector_type(8)))  float  v8f;

union Frag16 { v16bf v; uint4 u[2]; };
union Oct    { __bf16 t[8]; uint4 u; };

__device__ __forceinline__ v8f wmma_bf16(v16bf a, v16bf b, v8f c) {
  // (neg_a, A, neg_b, B, c_mod, C, reuse_a, reuse_b)
  return __builtin_amdgcn_wmma_f32_16x16x32_bf16(false, a, false, b,
                                                 (short)0, c, false, false);
}

// ---- gfx1250 async-to-LDS path (guarded) ----------------------------------
#if defined(__has_builtin)
#  if __has_builtin(__builtin_amdgcn_global_load_async_to_lds_b128) && \
      __has_builtin(__builtin_amdgcn_s_wait_asynccnt)
#    define ATHENA_ASYNC 1
#  endif
#endif
#ifndef ATHENA_ASYNC
#  define ATHENA_ASYNC 0
#endif

#if ATHENA_ASYNC
// Builtin signature (from clang diagnostic): param0 = AS1 (global/__device__)
// pointer to int __vector_size__(16); param1 = AS3 (LDS) pointer; then two
// integer immediates (offset, cpol).
typedef int v4i_raw __attribute__((__vector_size__(16)));
typedef __attribute__((address_space(1))) v4i_raw* gas_v4i;
typedef __attribute__((address_space(3))) v4i_raw* las_v4i;
#endif

__device__ __forceinline__ void stage_b128(const void* g, void* l) {
#if ATHENA_ASYNC
  __builtin_amdgcn_global_load_async_to_lds_b128((gas_v4i)g, (las_v4i)l, 0, 0);
#else
  *(uint4*)l = *(const uint4*)g;
#endif
}
__device__ __forceinline__ void wait_async_stage() {
#if ATHENA_ASYNC
  __builtin_amdgcn_s_wait_asynccnt(0);
#endif
}

constexpr int NROWS = 4096;   // B*S
constexpr int HDIM  = 2048;   // hidden
constexpr int SEQ   = 2048;
constexpr int HD    = 128;    // head dim

// ---------------------------------------------------------------------------
// GEMM: C[4096,2048] = A[4096,2048] x W[2048,2048] + bias
// Block tile 128x128, BK=32, 256 threads (8 waves). Wave grid 4(M) x 2(N),
// each wave computes 2x4 = 8 WMMA 16x16 tiles.
// ---------------------------------------------------------------------------
template<bool A_BF16, bool OUT_BF16>
__global__ __launch_bounds__(256)
void gemm_wmma(const void* __restrict__ Ap, const float* __restrict__ W,
               const float* __restrict__ bias, void* __restrict__ Cp)
{
  constexpr int N = HDIM, K = HDIM;
  __shared__ alignas(16) __bf16 sA[128 * 32];   // [m][k]
  __shared__ alignas(16) __bf16 sB[128 * 32];   // [n][k] (transposed tile of W)

  const int tid  = threadIdx.x;
  const int wave = tid >> 5, lane = tid & 31;
  const int h    = lane >> 4, l16 = lane & 15;
  const int wm   = wave & 3,  wn  = wave >> 2;
  const int bm   = blockIdx.x, bn = blockIdx.y;

  const float*  Af = (const float*)Ap;
  const __bf16* Ab = (const __bf16*)Ap;

  v8f acc[2][4] = {};

  for (int kb = 0; kb < K; kb += 32) {
    __syncthreads();
    // ---- stage A tile (convert fp32->bf16 if needed) -------------------
#pragma unroll
    for (int i = 0; i < 2; ++i) {
      int o   = tid * 2 + i;          // 0..511 octets (8 bf16 each)
      int row = o >> 2, q = o & 3;    // col = q*8
      size_t gidx = (size_t)(bm * 128 + row) * K + kb + q * 8;
      if constexpr (A_BF16) {
        stage_b128(Ab + gidx, &sA[row * 32 + q * 8]);    // async-to-LDS
      } else {
        const float* src = Af + gidx;
        if (kb + 32 < K) __builtin_prefetch(src + 32, 0, 1);
        float4 a0 = *(const float4*)src;
        float4 a1 = *(const float4*)(src + 4);
        Oct t;
        t.t[0] = (__bf16)a0.x; t.t[1] = (__bf16)a0.y;
        t.t[2] = (__bf16)a0.z; t.t[3] = (__bf16)a0.w;
        t.t[4] = (__bf16)a1.x; t.t[5] = (__bf16)a1.y;
        t.t[6] = (__bf16)a1.z; t.t[7] = (__bf16)a1.w;
        *(uint4*)&sA[row * 32 + q * 8] = t.u;
      }
    }
    // ---- stage B tile transposed: sB[n][k] = W[kb+k][bn*128+n] ---------
    {
      int k  = tid >> 3;              // 0..31
      int n0 = (tid & 7) * 16;        // 0..112
      const float* src = W + (size_t)(kb + k) * N + bn * 128 + n0;
      if (kb + 32 < K) __builtin_prefetch(src + (size_t)32 * N, 0, 1);
#pragma unroll
      for (int j = 0; j < 16; ++j)
        sB[(n0 + j) * 32 + k] = (__bf16)src[j];
    }
    if constexpr (A_BF16) wait_async_stage();
    __syncthreads();

    // ---- fragments + WMMA ---------------------------------------------
    Frag16 af[2], bfm[4];
#pragma unroll
    for (int mt = 0; mt < 2; ++mt) {
      const __bf16* p = &sA[(wm * 32 + mt * 16 + l16) * 32];
      af[mt].u[0] = *(const uint4*)(p + h * 8);        // K octets 0/1
      af[mt].u[1] = *(const uint4*)(p + 16 + h * 8);   // K octets 2/3
    }
#pragma unroll
    for (int nt = 0; nt < 4; ++nt) {
      const __bf16* p = &sB[(wn * 64 + nt * 16 + l16) * 32];
      bfm[nt].u[0] = *(const uint4*)(p + h * 8);
      bfm[nt].u[1] = *(const uint4*)(p + 16 + h * 8);
    }
#pragma unroll
    for (int mt = 0; mt < 2; ++mt)
#pragma unroll
      for (int nt = 0; nt < 4; ++nt)
        acc[mt][nt] = wmma_bf16(af[mt].v, bfm[nt].v, acc[mt][nt]);
  }

  // ---- epilogue: bias add + store --------------------------------------
  float*  Cf = (float*)Cp;
  __bf16* Cb = (__bf16*)Cp;
#pragma unroll
  for (int mt = 0; mt < 2; ++mt)
#pragma unroll
    for (int nt = 0; nt < 4; ++nt) {
      int col  = bn * 128 + wn * 64 + nt * 16 + l16;
      float bv = bias[col];
#pragma unroll
      for (int r = 0; r < 8; ++r) {
        int row = bm * 128 + wm * 32 + mt * 16 + h * 8 + r;
        float v = acc[mt][nt][r] + bv;
        if constexpr (OUT_BF16) Cb[(size_t)row * N + col] = (__bf16)v;
        else                    Cf[(size_t)row * N + col] = v;
      }
    }
}

// ---------------------------------------------------------------------------
// Flash attention: one WG per (128-query block, head, batch). 8 waves, each
// wave owns one 16-row m-tile. KV streamed in 64-row tiles through LDS
// (async b128 staging), V transposed LDS->LDS for the PV B-fragment layout.
// ---------------------------------------------------------------------------
__device__ __forceinline__ float rowred_max(float v) {
#pragma unroll
  for (int m = 1; m < 16; m <<= 1) v = fmaxf(v, __shfl_xor(v, m, 32));
  return v;
}
__device__ __forceinline__ float rowred_sum(float v) {
#pragma unroll
  for (int m = 1; m < 16; m <<= 1) v += __shfl_xor(v, m, 32);
  return v;
}

__global__ __launch_bounds__(256)
void attn_wmma(const __bf16* __restrict__ Q, const __bf16* __restrict__ Km,
               const __bf16* __restrict__ V, __bf16* __restrict__ Ctx)
{
  __shared__ alignas(16) __bf16 sK[64 * 128];    // [key][d]  (== B^T for QK^T)
  __shared__ alignas(16) __bf16 sVr[64 * 128];   // [key][d]  row-major V stage
  __shared__ alignas(16) __bf16 sVt[128 * 64];   // [d][key]  (== B^T for P*V)
  __shared__ alignas(16) __bf16 sP[8][16 * 64];  // per-wave P tile

  const int tid  = threadIdx.x;
  const int wave = tid >> 5, lane = tid & 31;
  const int h    = lane >> 4, l16 = lane & 15;
  const int qblk = blockIdx.x, head = blockIdx.y, b = blockIdx.z;
  const int rowbase = b * SEQ;
  const int colbase = head * HD;
  const int qrow    = rowbase + qblk * 128 + wave * 16 + l16;

  // Q fragments: 16x128 in 4 K=32 chunks, loaded once (A layout)
  Frag16 qf[4];
#pragma unroll
  for (int kc = 0; kc < 4; ++kc) {
    const __bf16* p = Q + (size_t)qrow * HDIM + colbase + kc * 32;
    qf[kc].u[0] = *(const uint4*)(p + h * 8);
    qf[kc].u[1] = *(const uint4*)(p + 16 + h * 8);
  }

  v8f o[8] = {};
  float mrow[8], lrow[8];
#pragma unroll
  for (int r = 0; r < 8; ++r) { mrow[r] = -__builtin_inff(); lrow[r] = 0.f; }
  const float scale = 0.08838834764831845f;  // 1/sqrt(128)

  for (int kv = 0; kv < SEQ; kv += 64) {
    __syncthreads();   // previous compute / transpose done with sK,sVr,sVt
    // ---- stage K and V tiles row-major (async b128 -> LDS) ------------
#pragma unroll
    for (int i = 0; i < 4; ++i) {
      int c = tid * 4 + i;            // 0..1023 octets
      int row = c >> 4, q = c & 15;
      size_t gidx = (size_t)(rowbase + kv + row) * HDIM + colbase + q * 8;
      stage_b128(Km + gidx, &sK[row * 128 + q * 8]);
      stage_b128(V  + gidx, &sVr[row * 128 + q * 8]);
    }
    wait_async_stage();
    __syncthreads();
    // ---- LDS->LDS transpose: sVt[d][key] = sVr[key][d] ----------------
#pragma unroll
    for (int i = 0; i < 4; ++i) {
      int c = tid * 4 + i;
      int key = c >> 4, dg = c & 15;  // 8 consecutive d per octet
      Oct t;
      t.u = *(const uint4*)&sVr[key * 128 + dg * 8];
#pragma unroll
      for (int j = 0; j < 8; ++j)
        sVt[(dg * 8 + j) * 64 + key] = t.t[j];
    }
    __syncthreads();

    // ---- S = Q K^T  (4 n-tiles x 4 k-chunks) --------------------------
    v8f sacc[4] = {};
#pragma unroll
    for (int nt = 0; nt < 4; ++nt)
#pragma unroll
      for (int kc = 0; kc < 4; ++kc) {
        Frag16 kf;
        const __bf16* p = &sK[(nt * 16 + l16) * 128 + kc * 32];
        kf.u[0] = *(const uint4*)(p + h * 8);
        kf.u[1] = *(const uint4*)(p + 16 + h * 8);
        sacc[nt] = wmma_bf16(qf[kc].v, kf.v, sacc[nt]);
      }

    // ---- online softmax ----------------------------------------------
    float mnew[8], alpha[8], rsum[8];
#pragma unroll
    for (int r = 0; r < 8; ++r) {
      float v = fmaxf(fmaxf(sacc[0][r], sacc[1][r]),
                      fmaxf(sacc[2][r], sacc[3][r]));
      v = rowred_max(v) * scale;
      mnew[r]  = fmaxf(mrow[r], v);
      alpha[r] = __expf(mrow[r] - mnew[r]);
      rsum[r]  = 0.f;
    }
#pragma unroll
    for (int nt = 0; nt < 4; ++nt)
#pragma unroll
      for (int r = 0; r < 8; ++r) {
        float p = __expf(sacc[nt][r] * scale - mnew[r]);
        rsum[r] += p;
        sP[wave][(h * 8 + r) * 64 + nt * 16 + l16] = (__bf16)p;  // C -> LDS
      }
#pragma unroll
    for (int r = 0; r < 8; ++r) {
      rsum[r] = rowred_sum(rsum[r]);
      lrow[r] = lrow[r] * alpha[r] + rsum[r];
      mrow[r] = mnew[r];
    }
#pragma unroll
    for (int dt = 0; dt < 8; ++dt)
#pragma unroll
      for (int r = 0; r < 8; ++r) o[dt][r] *= alpha[r];

    // ---- O += P V  (A layout of P from per-wave LDS tile) -------------
    Frag16 pf[2];
#pragma unroll
    for (int kc = 0; kc < 2; ++kc) {
      const __bf16* p = &sP[wave][l16 * 64 + kc * 32];
      pf[kc].u[0] = *(const uint4*)(p + h * 8);
      pf[kc].u[1] = *(const uint4*)(p + 16 + h * 8);
    }
#pragma unroll
    for (int dt = 0; dt < 8; ++dt)
#pragma unroll
      for (int kc = 0; kc < 2; ++kc) {
        Frag16 vf;
        const __bf16* p = &sVt[(dt * 16 + l16) * 64 + kc * 32];
        vf.u[0] = *(const uint4*)(p + h * 8);
        vf.u[1] = *(const uint4*)(p + 16 + h * 8);
        o[dt] = wmma_bf16(pf[kc].v, vf.v, o[dt]);
      }
  }

  // ---- finalize: O /= l, store bf16 context ----------------------------
#pragma unroll
  for (int r = 0; r < 8; ++r) lrow[r] = 1.f / lrow[r];
#pragma unroll
  for (int dt = 0; dt < 8; ++dt)
#pragma unroll
    for (int r = 0; r < 8; ++r) {
      int row = rowbase + qblk * 128 + wave * 16 + h * 8 + r;
      int col = colbase + dt * 16 + l16;
      Ctx[(size_t)row * HDIM + col] = (__bf16)(o[dt][r] * lrow[r]);
    }
}

// ---------------------------------------------------------------------------
extern "C" void kernel_launch(void* const* d_in, const int* in_sizes, int n_in,
                              void* d_out, int out_size, void* d_ws, size_t ws_size,
                              hipStream_t stream) {
  const float* X  = (const float*)d_in[0];
  const float* Wq = (const float*)d_in[1];
  const float* bq = (const float*)d_in[2];
  const float* Wk = (const float*)d_in[3];
  const float* bk = (const float*)d_in[4];
  const float* Wv = (const float*)d_in[5];
  const float* bv = (const float*)d_in[6];
  const float* Wo = (const float*)d_in[7];
  const float* bo = (const float*)d_in[8];

  // Workspace: 4 bf16 buffers of 4096x2048 (16 MiB each, 64 MiB total)
  const size_t elems = (size_t)NROWS * HDIM;
  __bf16* Qw = (__bf16*)d_ws;
  __bf16* Kw = Qw + elems;
  __bf16* Vw = Kw + elems;
  __bf16* Cw = Vw + elems;

  dim3 blk(256);
  dim3 gg(NROWS / 128, HDIM / 128);        // 32 x 16

  gemm_wmma<false, true><<<gg, blk, 0, stream>>>((const void*)X, Wq, bq, (void*)Qw);
  gemm_wmma<false, true><<<gg, blk, 0, stream>>>((const void*)X, Wk, bk, (void*)Kw);
  gemm_wmma<false, true><<<gg, blk, 0, stream>>>((const void*)X, Wv, bv, (void*)Vw);

  dim3 ga(SEQ / 128, HDIM / HD, 2);        // 16 q-blocks x 16 heads x 2 batch
  attn_wmma<<<ga, blk, 0, stream>>>(Qw, Kw, Vw, Cw);

  gemm_wmma<true, false><<<gg, blk, 0, stream>>>((const void*)Cw, Wo, bo, d_out);
}
